// WaveletFreqWiseAttention_9603546874547
// MI455X (gfx1250) — matrix operand order repaired and checked by reading, not compiled, not run on hardware
//
#include <hip/hip_runtime.h>
#include <hip/hip_bf16.h>
#include <math.h>

// ---------------- problem constants ----------------
#define B_   2
#define C_   128
#define H_   256
#define W_   256
#define NH_  8
#define HW_  (H_ * W_)        // 65536
#define Hh_  (H_ / 2)         // 128
#define Wh_  (W_ / 2)         // 128
#define hw_  (Hh_ * Wh_)      // 16384

typedef __attribute__((ext_vector_type(16))) _Float16 v16h;
typedef __attribute__((ext_vector_type(8)))  float    v8f;
typedef __attribute__((ext_vector_type(4)))  float    v4f;

__device__ __forceinline__ float sigmoidf_(float x) { return 1.0f / (1.0f + expf(-x)); }

// Build the A-operand f16 vector from one contiguous 16-float run.
// A-matrix (16x32 f16): lane row = lane&15; element j -> K = 16*half + j
// (contiguous order, matching the B operands).
__device__ __forceinline__ v16h load_a_f32(const float* __restrict__ row, int half) {
  v4f w0 = *(const v4f*)(row + 16 * half);
  v4f w1 = *(const v4f*)(row + 16 * half + 4);
  v4f w2 = *(const v4f*)(row + 16 * half + 8);
  v4f w3 = *(const v4f*)(row + 16 * half + 12);
  v16h a;
#pragma unroll
  for (int j = 0; j < 4; ++j) {
    a[j]      = (_Float16)w0[j];
    a[j + 4]  = (_Float16)w1[j];
    a[j + 8]  = (_Float16)w2[j];
    a[j + 12] = (_Float16)w3[j];
  }
  return a;
}

// B-operand when K runs contiguously in memory (Gram kernel):
// element j -> K = 16*half + j (16 consecutive floats).
__device__ __forceinline__ v16h load_b_contig_f32(const float* __restrict__ row, int half) {
  v4f b0 = *(const v4f*)(row + 16 * half);
  v4f b1 = *(const v4f*)(row + 16 * half + 4);
  v4f b2 = *(const v4f*)(row + 16 * half + 8);
  v4f b3 = *(const v4f*)(row + 16 * half + 12);
  v16h b;
#pragma unroll
  for (int j = 0; j < 4; ++j) {
    b[j]      = (_Float16)b0[j];
    b[j + 4]  = (_Float16)b1[j];
    b[j + 8]  = (_Float16)b2[j];
    b[j + 12] = (_Float16)b3[j];
  }
  return b;
}

// =====================================================================
// 1x1-conv GEMM:  Y[b, m, n] = sum_k W[m, k] * X[b, in_c0+k, n] + bias[m]
// K is a template parameter -> fully unrolled WMMA chain.
// Each wave computes a 16(out-ch) x 64(pixel) register block: the A tile is
// loaded once per k-step and reused by 4 v_wmma_f32_16x16x32_f16 ops.
// grid = (Npix/256, M/16, B), block = 128 (4 waves, each 4 pixel tiles).
// act: 0 = none, 1 = sigmoid.
// =====================================================================
template <int K>
__global__ void gemm1x1_wmma(const float* __restrict__ X, const float* __restrict__ W,
                             const float* __restrict__ bias, float* __restrict__ Y,
                             int Npix, long in_bstride, long out_bstride,
                             int in_c0, int act) {
  const int lane = threadIdx.x & 31;
  const int wid  = threadIdx.x >> 5;
  const int half = lane >> 4;       // 0: lanes 0-15, 1: lanes 16-31
  const int l    = lane & 15;
  const int n0 = blockIdx.x * 256 + wid * 64;     // wave's 64-pixel span
  const int m0 = blockIdx.y * 16;
  const int b  = blockIdx.z;

  const float* Xb = X + (long)b * in_bstride + (long)in_c0 * Npix + n0 + l;
  const float* Wr = W + (long)(m0 + l) * K;

  v8f acc0 = {}, acc1 = {}, acc2 = {}, acc3 = {};
#pragma unroll
  for (int k0 = 0; k0 < K; k0 += 32) {
    v16h a = load_a_f32(Wr + k0, half);           // shared across 4 tiles
    const float* Xk = Xb + (long)(k0 + 16 * half) * Npix;
    v16h bm0, bm1, bm2, bm3;
#pragma unroll
    for (int j = 0; j < 16; ++j) {
      // B-matrix (32x16 f16): lanes 0-15 K=0..15, lanes 16-31 K=16..31;
      // K dimension = input channels (stride Npix in memory).
      const float* xp = Xk + (long)j * Npix;
      bm0[j] = (_Float16)xp[0];
      bm1[j] = (_Float16)xp[16];
      bm2[j] = (_Float16)xp[32];
      bm3[j] = (_Float16)xp[48];
    }
    acc0 = __builtin_amdgcn_wmma_f32_16x16x32_f16(false, a, false, bm0, (short)0, acc0, false, false);
    acc1 = __builtin_amdgcn_wmma_f32_16x16x32_f16(false, a, false, bm1, (short)0, acc1, false, false);
    acc2 = __builtin_amdgcn_wmma_f32_16x16x32_f16(false, a, false, bm2, (short)0, acc2, false, false);
    acc3 = __builtin_amdgcn_wmma_f32_16x16x32_f16(false, a, false, bm3, (short)0, acc3, false, false);
  }
#pragma unroll
  for (int r = 0; r < 8; ++r) {
    int row = m0 + r + 8 * half;                  // C/D layout: VGPR r -> M=r / M=8+r
    float bsv = bias[row];
    float* yp = Y + (long)b * out_bstride + (long)row * Npix + n0 + l;
    float v0 = acc0[r] + bsv, v1 = acc1[r] + bsv, v2 = acc2[r] + bsv, v3 = acc3[r] + bsv;
    if (act) { v0 = sigmoidf_(v0); v1 = sigmoidf_(v1); v2 = sigmoidf_(v2); v3 = sigmoidf_(v3); }
    yp[0] = v0; yp[16] = v1; yp[32] = v2; yp[48] = v3;
  }
}

// =====================================================================
// Channel-attention Gram matrix: gram[bh][c][d] += sum_n q[c,n]*k[d,n]
// Both operands contiguous along K=n -> all-b128 loads. Split-K over
// blockIdx.x with f32 atomic reduction of the 16x16 tile.
// grid = (Npix/chunk, B*NH), block = 32 (one wave).
// =====================================================================
__global__ void gram_wmma(const float* __restrict__ q, const float* __restrict__ k,
                          float* __restrict__ gram, int Npix, int chunk) {
  const int lane = threadIdx.x & 31;
  const int half = lane >> 4, l = lane & 15;
  const int bh = blockIdx.y;
  const int b = bh / NH_, h = bh % NH_;
  const float* qb = q + ((long)b * C_ + h * 16 + l) * (long)Npix;
  const float* kb = k + ((long)b * C_ + h * 16 + l) * (long)Npix;

  v8f acc = {};
  const int n_start = blockIdx.x * chunk;
#pragma unroll 4
  for (int n0 = n_start; n0 < n_start + chunk; n0 += 32) {
    __builtin_prefetch(qb + n0 + 256, 0, 3);      // global_prefetch 8 k-tiles (256 floats) ahead
    __builtin_prefetch(kb + n0 + 256, 0, 3);
    v16h a  = load_a_f32(qb + n0, half);          // A = q (M=c, K=n)
    v16h bm = load_b_contig_f32(kb + n0, half);   // B = k^T (K=n, N=d)
    acc = __builtin_amdgcn_wmma_f32_16x16x32_f16(false, a, false, bm, (short)0, acc, false, false);
  }
#pragma unroll
  for (int r = 0; r < 8; ++r)
    atomicAdd(&gram[((long)bh * 16 + r + 8 * half) * 16 + l], acc[r]);
}

// ---------------- depthwise conv (zero pad, any kh x kw) ----------------
__global__ void dwconv(const float* __restrict__ in, const float* __restrict__ w,
                       float* __restrict__ out, int Hx, int Wx, int kh, int kw,
                       long ibs, long obs, int ic0) {
  const int c = blockIdx.y, b = blockIdx.z;
  const int p = blockIdx.x * 256 + threadIdx.x;
  const int y = p / Wx, x = p % Wx;
  const float* ib = in + (long)b * ibs + (long)(ic0 + c) * Hx * Wx;
  const float* wc = w + (long)c * kh * kw;
  float s = 0.f;
  for (int i = 0; i < kh; ++i) {
    int yy = y + i - kh / 2;
    if (yy < 0 || yy >= Hx) continue;
    for (int j = 0; j < kw; ++j) {
      int xx = x + j - kw / 2;
      if (xx < 0 || xx >= Wx) continue;
      s += ib[(long)yy * Wx + xx] * wc[i * kw + j];
    }
  }
  out[(long)b * obs + (long)c * Hx * Wx + p] = s;
}

// ---------------- fused gated depthwise: out = conv(x,w0)*sigmoid(conv(x,w1)) ----
__global__ void dwconv_gated(const float* __restrict__ in, const float* __restrict__ w,
                             float* __restrict__ out, int Hx, int Wx, int kh, int kw,
                             long ibs, long obs, int oc0) {
  const int c = blockIdx.y, b = blockIdx.z;
  const int p = blockIdx.x * 256 + threadIdx.x;
  const int y = p / Wx, x = p % Wx;
  const float* ib = in + (long)b * ibs + (long)c * Hx * Wx;
  const float* w0 = w + (long)c * kh * kw;
  const float* w1 = w + (long)(C_ + c) * kh * kw;
  float s0 = 0.f, s1 = 0.f;
  for (int i = 0; i < kh; ++i) {
    int yy = y + i - kh / 2;
    if (yy < 0 || yy >= Hx) continue;
    for (int j = 0; j < kw; ++j) {
      int xx = x + j - kw / 2;
      if (xx < 0 || xx >= Wx) continue;
      float v = ib[(long)yy * Wx + xx];
      s0 += v * w0[i * kw + j];
      s1 += v * w1[i * kw + j];
    }
  }
  out[(long)b * obs + (long)(oc0 + c) * Hx * Wx + p] = s0 * sigmoidf_(s1);
}

// ---------------- Haar DWT / fused masked IDWT ----------------
__global__ void dwt_haar(const float* __restrict__ in, float* __restrict__ ll,
                         float* __restrict__ lh, float* __restrict__ hl,
                         float* __restrict__ hh) {
  const int c = blockIdx.y, b = blockIdx.z;
  const int p = blockIdx.x * 256 + threadIdx.x;
  const int oy = p >> 7, ox = p & (Wh_ - 1);
  const float* ib = in + ((long)b * C_ + c) * (long)HW_;
  float a  = ib[(2 * oy) * W_ + 2 * ox];
  float bb = ib[(2 * oy) * W_ + 2 * ox + 1];
  float cc = ib[(2 * oy + 1) * W_ + 2 * ox];
  float dd = ib[(2 * oy + 1) * W_ + 2 * ox + 1];
  long o = ((long)b * C_ + c) * (long)hw_ + p;
  ll[o] = (a + bb + cc + dd) * 0.5f;
  lh[o] = (a + bb - cc - dd) * 0.5f;
  hl[o] = (a - bb + cc - dd) * 0.5f;
  hh[o] = (a - bb - cc + dd) * 0.5f;
}

__global__ void idwt_haar(const float* __restrict__ qll, const float* __restrict__ glh,
                          const float* __restrict__ ghl, const float* __restrict__ ghh,
                          const float* __restrict__ masks, float* __restrict__ out) {
  const int c = blockIdx.y, b = blockIdx.z;
  const int p = blockIdx.x * 256 + threadIdx.x;
  const int oy = p >> 7, ox = p & (Wh_ - 1);
  long o  = ((long)b * C_ + c) * (long)hw_ + p;
  long mb = (long)b * 5 * C_ * hw_ + (long)c * hw_ + p;
  float ll = qll[o];
  float lh = glh[o] * masks[mb + (long)1 * C_ * hw_];
  float hl = ghl[o] * masks[mb + (long)2 * C_ * hw_];
  float hh = ghh[o] * masks[mb + (long)3 * C_ * hw_];
  float a  = (ll + lh + hl + hh) * 0.5f;
  float bb = (ll + lh - hl - hh) * 0.5f;
  float cc = (ll - lh + hl - hh) * 0.5f;
  float dd = (ll - lh - hl + hh) * 0.5f;
  float* ob = out + ((long)b * C_ + c) * (long)HW_;
  ob[(2 * oy) * W_ + 2 * ox]         = a;
  ob[(2 * oy) * W_ + 2 * ox + 1]     = bb;
  ob[(2 * oy + 1) * W_ + 2 * ox]     = cc;
  ob[(2 * oy + 1) * W_ + 2 * ox + 1] = dd;
}

// ---------------- out += add * mask (batch strides differ per tensor) ----------
__global__ void residual_mask(float* __restrict__ out, const float* __restrict__ add,
                              const float* __restrict__ mask,
                              long per, long obs, long abs_, long mbs) {
  long gid = (long)blockIdx.x * 256 + threadIdx.x;
  long b = gid / per, r = gid % per;
  out[b * obs + r] += add[b * abs_ + r] * mask[b * mbs + r];
}

__global__ void fillz(float* __restrict__ p, long n) {
  long i = (long)blockIdx.x * 256 + threadIdx.x;
  if (i < n) p[i] = 0.f;
}

// ---------------- row L2 norm (rows contiguous, length n) ----------------
__global__ void rownorm(const float* __restrict__ t, float* __restrict__ o, int n) {
  __shared__ float red[256];
  const float* base = t + (long)blockIdx.x * n;
  float s = 0.f;
  for (int i = threadIdx.x; i < n; i += 256) { float v = base[i]; s += v * v; }
  red[threadIdx.x] = s; __syncthreads();
  for (int st = 128; st > 0; st >>= 1) {
    if (threadIdx.x < st) red[threadIdx.x] += red[threadIdx.x + st];
    __syncthreads();
  }
  if (threadIdx.x == 0) o[blockIdx.x] = fmaxf(sqrtf(red[0]), 1e-12f);
}

// ---------------- scale by 1/(|q||k|)*temp, softmax over d (16) ------------
__global__ void softmax16(float* __restrict__ gram, const float* __restrict__ norms,
                          const float* __restrict__ temp) {
  const int tid = threadIdx.x;              // 256 = B*NH*16 rows
  const int b = tid >> 7, rem = tid & 127, h = rem >> 4, c = rem & 15;
  const float nq = norms[b * C_ + h * 16 + c];
  const float t  = temp[h];
  float vals[16];
  long base = ((long)(b * NH_ + h) * 16 + c) * 16;
  float mx = -1e30f;
#pragma unroll
  for (int d = 0; d < 16; ++d) {
    float nk = norms[B_ * C_ + b * C_ + h * 16 + d];
    vals[d] = gram[base + d] / (nq * nk) * t;
    mx = fmaxf(mx, vals[d]);
  }
  float sum = 0.f;
#pragma unroll
  for (int d = 0; d < 16; ++d) { vals[d] = expf(vals[d] - mx); sum += vals[d]; }
  float inv = 1.f / sum;
#pragma unroll
  for (int d = 0; d < 16; ++d) gram[base + d] = vals[d] * inv;
}

// ---------------- out[b,h*16+c,n] = sum_d att[c,d]*v[b,h*16+d,n] ------------
__global__ void attv(const float* __restrict__ att, const float* __restrict__ v,
                     float* __restrict__ out) {
  __shared__ float s[256];
  const int bh = blockIdx.y, b = bh / NH_, h = bh % NH_;
  const int tid = threadIdx.x;
  s[tid] = att[(long)bh * 256 + tid];
  __syncthreads();
  long n = (long)blockIdx.x * 256 + tid;
  const float* vb = v + ((long)b * C_ + h * 16) * (long)HW_ + n;
  float vv[16];
#pragma unroll
  for (int d = 0; d < 16; ++d) vv[d] = vb[(long)d * HW_];
  float* ob = out + ((long)b * C_ + h * 16) * (long)HW_ + n;
#pragma unroll
  for (int c = 0; c < 16; ++c) {
    float sum = 0.f;
#pragma unroll
    for (int d = 0; d < 16; ++d) sum += s[c * 16 + d] * vv[d];
    ob[(long)c * HW_] = sum;
  }
}

// =====================================================================
extern "C" void kernel_launch(void* const* d_in, const int* in_sizes, int n_in,
                              void* d_out, int out_size, void* d_ws, size_t ws_size,
                              hipStream_t stream) {
  const float* x       = (const float*)d_in[0];
  const float* prior   = (const float*)d_in[1];
  const float* w_lin   = (const float*)d_in[2];
  const float* b_lin   = (const float*)d_in[3];
  const float* w_qd    = (const float*)d_in[4];
  const float* w_kd    = (const float*)d_in[5];
  const float* w_vd    = (const float*)d_in[6];
  const float* g_llr   = (const float*)d_in[7];
  const float* g_llh   = (const float*)d_in[8];
  const float* g_lh    = (const float*)d_in[9];
  const float* g_hl    = (const float*)d_in[10];
  const float* g_hh    = (const float*)d_in[11];
  const float* w_prior = (const float*)d_in[12];
  const float* b_prior = (const float*)d_in[13];
  const float* w_pw1   = (const float*)d_in[14];
  const float* b_pw1   = (const float*)d_in[15];
  const float* w_pw2   = (const float*)d_in[16];
  const float* b_pw2   = (const float*)d_in[17];
  const float* w_cat   = (const float*)d_in[18];
  const float* b_cat   = (const float*)d_in[19];
  const float* temp    = (const float*)d_in[20];
  const float* w_out   = (const float*)d_in[21];
  const float* b_out   = (const float*)d_in[22];
  float* out = (float*)d_out;
  float* ws  = (float*)d_ws;

  const size_t S2 = (size_t)B_ * C_ * HW_;   // 16.8M floats
  const size_t S3 = (size_t)B_ * C_ * hw_;   //  4.2M floats

  // qkv occupies [0, 3*S2) early; q_rec/att_out reuse it after the dwconvs.
  float* qkv    = ws;
  float* qrec   = ws;
  float* attout = ws + S2;
  size_t base = 3 * S2;
  float* qd    = ws + base; base += S2;
  float* kd    = ws + base; base += S2;
  float* vd    = ws + base; base += S2;
  float* llb   = ws + base; base += S3;
  float* lhb   = ws + base; base += S3;
  float* hlb   = ws + base; base += S3;
  float* hhb   = ws + base; base += S3;
  float* tmp   = ws + base; base += S3;
  float* glh   = ws + base; base += S3;
  float* ghl   = ws + base; base += S3;
  float* ghh   = ws + base; base += S3;
  float* cat   = ws + base; base += 2 * S3;          // [llr | llh] per batch
  float* masks = ws + base; base += 5 * S3;
  float* pw    = ws + base; base += S3;
  float* qll   = ws + base; base += S3;
  float* gram  = ws + base; base += (size_t)B_ * NH_ * 256;
  float* norms = ws + base; base += 2 * (size_t)B_ * C_;

  // 1) qkv = W_lin @ x + b  (WMMA GEMM, M=384, K=128)
  gemm1x1_wmma<C_><<<dim3(HW_ / 256, 3 * C_ / 16, B_), 128, 0, stream>>>(
      x, w_lin, b_lin, qkv, HW_, (long)C_ * HW_, (long)3 * C_ * HW_, 0, 0);

  // 2) depthwise 3x3 on q,k,v
  dim3 gdw(HW_ / 256, C_, B_);
  dwconv<<<gdw, 256, 0, stream>>>(qkv, w_qd, qd, H_, W_, 3, 3, (long)3 * C_ * HW_, (long)C_ * HW_, 0);
  dwconv<<<gdw, 256, 0, stream>>>(qkv, w_kd, kd, H_, W_, 3, 3, (long)3 * C_ * HW_, (long)C_ * HW_, C_);
  dwconv<<<gdw, 256, 0, stream>>>(qkv, w_vd, vd, H_, W_, 3, 3, (long)3 * C_ * HW_, (long)C_ * HW_, 2 * C_);

  // 3) Haar DWT of q
  dim3 gsb(hw_ / 256, C_, B_);
  dwt_haar<<<gsb, 256, 0, stream>>>(qd, llb, lhb, hlb, hhb);

  // 4) gated depthwise convs
  const long bs3 = (long)C_ * hw_;
  dwconv_gated<<<gsb, 256, 0, stream>>>(llb, g_llr,              tmp, Hh_, Wh_, 3, 3, bs3, bs3, 0);
  dwconv_gated<<<gsb, 256, 0, stream>>>(tmp, g_llr + 2 * C_ * 9, cat, Hh_, Wh_, 3, 3, bs3, 2 * bs3, 0);
  dwconv_gated<<<gsb, 256, 0, stream>>>(llb, g_llh,              tmp, Hh_, Wh_, 3, 3, bs3, bs3, 0);
  dwconv_gated<<<gsb, 256, 0, stream>>>(tmp, g_llh + 2 * C_ * 9, cat, Hh_, Wh_, 3, 3, bs3, 2 * bs3, C_);
  dwconv_gated<<<gsb, 256, 0, stream>>>(lhb, g_lh, glh, Hh_, Wh_, 3, 5, bs3, bs3, 0);
  dwconv_gated<<<gsb, 256, 0, stream>>>(hlb, g_hl, ghl, Hh_, Wh_, 5, 3, bs3, bs3, 0);
  dwconv_gated<<<gsb, 256, 0, stream>>>(hhb, g_hh, ghh, Hh_, Wh_, 3, 3, bs3, bs3, 0);

  // 5) masks = sigmoid(W_prior @ prior + b)  (WMMA GEMM, M=640, act=sigmoid)
  gemm1x1_wmma<C_><<<dim3(hw_ / 256, 5 * C_ / 16, B_), 128, 0, stream>>>(
      prior, w_prior, b_prior, masks, hw_, bs3, 5 * bs3, 0, 1);

  // 6) q_llr += pw1(q_llr) * m_ll ;  q_llh += pw2(q_llh) * m_hp
  dim3 ggm(hw_ / 256, C_ / 16, B_);
  gemm1x1_wmma<C_><<<ggm, 128, 0, stream>>>(cat, w_pw1, b_pw1, pw, hw_, 2 * bs3, bs3, 0, 0);
  residual_mask<<<(unsigned)(B_ * C_ * hw_ / 256), 256, 0, stream>>>(
      cat, pw, masks, bs3, 2 * bs3, bs3, 5 * bs3);
  gemm1x1_wmma<C_><<<ggm, 128, 0, stream>>>(cat, w_pw2, b_pw2, pw, hw_, 2 * bs3, bs3, C_, 0);
  residual_mask<<<(unsigned)(B_ * C_ * hw_ / 256), 256, 0, stream>>>(
      cat + (size_t)C_ * hw_, pw, masks + (size_t)4 * C_ * hw_, bs3, 2 * bs3, bs3, 5 * bs3);

  // 7) q_ll = W_cat @ [llr;llh] + b  (WMMA GEMM, K=256)
  gemm1x1_wmma<2 * C_><<<ggm, 128, 0, stream>>>(cat, w_cat, b_cat, qll, hw_, 2 * bs3, bs3, 0, 0);

  // 8) fused masked IDWT -> reconstructed q
  idwt_haar<<<gsb, 256, 0, stream>>>(qll, glh, ghl, ghh, masks, qrec);

  // 9) channel attention: norms, split-K WMMA Gram, softmax, att@V
  fillz<<<(B_ * NH_ * 256 + 255) / 256, 256, 0, stream>>>(gram, (long)B_ * NH_ * 256);
  rownorm<<<B_ * C_, 256, 0, stream>>>(qrec, norms, HW_);
  rownorm<<<B_ * C_, 256, 0, stream>>>(kd, norms + B_ * C_, HW_);
  gram_wmma<<<dim3(64, B_ * NH_), 32, 0, stream>>>(qrec, kd, gram, HW_, HW_ / 64);
  softmax16<<<1, 256, 0, stream>>>(gram, norms, temp);
  attv<<<dim3(HW_ / 256, B_ * NH_), 256, 0, stream>>>(gram, vd, attout);

  // 10) final projection straight into d_out (WMMA GEMM)
  gemm1x1_wmma<C_><<<dim3(HW_ / 256, C_ / 16, B_), 128, 0, stream>>>(
      attout, w_out, b_out, out, HW_, (long)C_ * HW_, (long)C_ * HW_, 0, 0);

  // 11) second tuple output: prior passthrough
  hipMemcpyAsync(out + S2, prior, (size_t)B_ * C_ * hw_ * sizeof(float),
                 hipMemcpyDeviceToDevice, stream);
}
